// CNNFocusing_52055003627669
// MI455X (gfx1250) — matrix operand order repaired and checked by reading, not compiled, not run on hardware
//
#include <hip/hip_runtime.h>
#include <hip/hip_bf16.h>

// MI455X (gfx1250): grouped CNN + landmark heads, all GEMMs via
// v_wmma_f32_16x16x32_f16 (f16 in, f32 accumulate), f16 intermediates.
//
// Activations are channel-innermost ([y][x][ci]) with im2col order
// K = (ky, kx, ci): every 8-wide K-run of a WMMA A-fragment is contiguous in
// LDS -> fragment = 2x ds_load_2addr_b64, branch-free. Padded K positions have
// zero weights, so A may read finite garbage there (in-bounds only).

typedef __attribute__((ext_vector_type(16))) _Float16 v16h;
typedef __attribute__((ext_vector_type(8)))  float    v8f;

#define G 72

struct Q4 { unsigned long long a, b, c, d; };   // 32B == v16h
struct H4 { _Float16 h[4]; };                   // 8B packed store

__device__ __forceinline__ unsigned long long ldu64(const _Float16* p) {
  return *(const unsigned long long*)p;
}
__device__ __forceinline__ v16h q4v(Q4 q) { return __builtin_bit_cast(v16h, q); }

// ---------------- weight prep: fp32 -> f16, transposed [co][Kpad] ----------------
// conv1: K = ky*32 + kx*4 + ci   (ci pad 3->4, per-ky pad 20->32; Kpad=160)
__global__ void prep_w1(const float* __restrict__ w1, _Float16* __restrict__ w1t) {
  int i = blockIdx.x * 256 + threadIdx.x;
  if (i >= G * 16 * 160) return;
  int g = i / 2560, rem = i % 2560, co = rem / 160, K = rem % 160;
  int ky = K >> 5, r = K & 31, kx = r >> 2, ci = r & 3;
  float v = 0.f;
  if (co < 8 && kx < 5 && ci < 3)
    v = w1[((g * 8 + co) * 3 + ci) * 25 + ky * 5 + kx];
  w1t[i] = (_Float16)v;
}
// conv2: K = ky*24 + kx*8 + ci   (72 valid, Kpad=96)
__global__ void prep_w2(const float* __restrict__ w2, _Float16* __restrict__ w2t) {
  int i = blockIdx.x * 256 + threadIdx.x;
  if (i >= G * 16 * 96) return;
  int g = i / 1536, rem = i % 1536, co = rem / 96, K = rem % 96;
  int ky = K / 24, r = K % 24, kx = r >> 3, ci = r & 7;
  float v = 0.f;
  if (ky < 3) v = w2[((g * 16 + co) * 8 + ci) * 9 + ky * 3 + kx];
  w2t[i] = (_Float16)v;
}
// conv3: K = ky*48 + kx*16 + ci  (144 valid, Kpad=160)
__global__ void prep_w3(const float* __restrict__ w3, _Float16* __restrict__ w3t) {
  int i = blockIdx.x * 256 + threadIdx.x;
  if (i >= G * 32 * 160) return;
  int g = i / 5120, rem = i % 5120, co = rem / 160, K = rem % 160;
  int ky = K / 48, r = K % 48, kx = r >> 4, ci = r & 15;
  float v = 0.f;
  if (ky < 3) v = w3[((g * 32 + co) * 16 + ci) * 9 + ky * 3 + kx];
  w3t[i] = (_Float16)v;
}
// heads: lw1T[g][n(64)][K(288)]
__global__ void prep_lw1(const float* __restrict__ lw1, _Float16* __restrict__ lw1t) {
  int i = blockIdx.x * 256 + threadIdx.x;
  if (i >= G * 64 * 288) return;
  int g = i / 18432, rem = i % 18432, n = rem / 288, K = rem % 288;
  lw1t[i] = (_Float16)lw1[((size_t)g * 288 + K) * 64 + n];
}

// ---------------- conv1 (Cin=3,k=5,Cout=8) + relu + maxpool2 ----------------
__global__ __launch_bounds__(256) void conv1_kernel(
    const float* __restrict__ x, const _Float16* __restrict__ w1t,
    const float* __restrict__ b1, _Float16* __restrict__ h1) {
  const int bg = blockIdx.x;
  const int b = bg / G, g = bg % G;
  const int tid = threadIdx.x, lane = tid & 31, wave = tid >> 5;
  const int col = lane & 15, half = lane >> 4, m = lane & 15;

  __shared__ _Float16 xs[46 * 46 * 4 + 16];   // [y][x][ci4] + zeroed slack
  __shared__ _Float16 wB[16 * 160];           // [co][K]
  __shared__ float scr[8][16][16];

  const float* xp = x + ((size_t)b * 216 + g * 3) * 2116;
  for (int p = tid; p < 2116; p += 256) {
    H4 h4;
    h4.h[0] = (_Float16)xp[p];
    h4.h[1] = (_Float16)xp[2116 + p];
    h4.h[2] = (_Float16)xp[4232 + p];
    h4.h[3] = (_Float16)0.f;
    ((unsigned long long*)xs)[p] = __builtin_bit_cast(unsigned long long, h4);
  }
  if (tid < 16) xs[46 * 46 * 4 + tid] = (_Float16)0.f;
  for (int i = tid; i < 640; i += 256)
    ((unsigned long long*)wB)[i] = ((const unsigned long long*)(w1t + g * 2560))[i];
  __syncthreads();

  v16h bf[5];
#pragma unroll
  for (int kc = 0; kc < 5; ++kc) {
    const _Float16* p = wB + col * 160 + kc * 32 + 8 * half;
    Q4 q; q.a = ldu64(p); q.b = ldu64(p + 4); q.c = ldu64(p + 16); q.d = ldu64(p + 20);
    bf[kc] = q4v(q);
  }

  const int pco = lane & 7, pxi = lane >> 3;           // pooling lane roles
  const float bias = b1[g * 8 + pco];
  _Float16* h1p = h1 + ((size_t)b * G + g) * 3528;     // [441][8]

  int ty = wave / 6, tc = wave % 6;
  for (int t = wave; t < 126; t += 8) {        // 21 row-pairs x 6 col-tiles
    int row = 2 * ty + (m >> 3);
    int cp = 8 * tc + (m & 7); if (cp > 41) cp = 41;
    v8f c = {};
#pragma unroll
    for (int kc = 0; kc < 5; ++kc) {           // ky == kc (32 K per ky)
      const _Float16* p = xs + ((row + kc) * 46 + cp) * 4 + 8 * half;
      Q4 q; q.a = ldu64(p); q.b = ldu64(p + 4); q.c = ldu64(p + 16); q.d = ldu64(p + 20);
      c = __builtin_amdgcn_wmma_f32_16x16x32_f16(false, q4v(q), false, bf[kc],
                                                 (short)0, c, false, false);
    }
#pragma unroll
    for (int r = 0; r < 8; ++r) scr[wave][r + 8 * half][col] = c[r];
    int px = 4 * tc + pxi;
    if (px < 21) {
      int m0 = 2 * pxi;
      float v0 = scr[wave][m0][pco],     v1 = scr[wave][m0 + 1][pco];
      float v2 = scr[wave][m0 + 8][pco], v3 = scr[wave][m0 + 9][pco];
      float mx = fmaxf(fmaxf(fmaxf(v0, v1), fmaxf(v2, v3)) + bias, 0.f);
      h1p[(ty * 21 + px) * 8 + pco] = (_Float16)mx;    // h1[b][g][y][x][8]
    }
    tc += 2; ty += 1; if (tc >= 6) { tc -= 6; ty += 1; }
  }
}

// ---------------- conv2 (Cin=8,k=3,Cout=16) + relu + maxpool2 ----------------
__global__ __launch_bounds__(256) void conv2_kernel(
    const _Float16* __restrict__ h1, const _Float16* __restrict__ w2t,
    const float* __restrict__ b2, _Float16* __restrict__ h2) {
  const int bg = blockIdx.x;
  const int b = bg / G, g = bg % G;
  const int tid = threadIdx.x, lane = tid & 31, wave = tid >> 5;
  const int col = lane & 15, half = lane >> 4, m = lane & 15;

  __shared__ _Float16 xs[21 * 21 * 8];        // [y][x][8] == h1 slice
  __shared__ _Float16 wB[16 * 96];
  __shared__ float scr[8][16][16];

  const unsigned long long* src = (const unsigned long long*)(h1 + ((size_t)b * G + g) * 3528);
  for (int i = tid; i < 441; i += 256) ((unsigned long long*)xs)[i] = src[i];
  for (int i = tid; i < 384; i += 256)
    ((unsigned long long*)wB)[i] = ((const unsigned long long*)(w2t + g * 1536))[i];
  __syncthreads();

  v16h bf[3];
  int ky0[3], of0[3], ky1[3], of1[3];
#pragma unroll
  for (int kc = 0; kc < 3; ++kc) {
    const _Float16* p = wB + col * 96 + kc * 32 + 8 * half;
    Q4 q; q.a = ldu64(p); q.b = ldu64(p + 4); q.c = ldu64(p + 16); q.d = ldu64(p + 20);
    bf[kc] = q4v(q);
    int K0 = kc * 32 + 8 * half;               // run starts, multiples of 8
    ky0[kc] = K0 / 24; of0[kc] = K0 - ky0[kc] * 24;
    int K1 = K0 + 16;
    ky1[kc] = K1 / 24; of1[kc] = K1 - ky1[kc] * 24;
  }

  const float bias = b2[g * 16 + col];         // pooling co == lane&15
  _Float16* h2p = h2 + ((size_t)b * G + g) * 1296;     // [81][16]

  int ty = wave / 3, tc = wave % 3;
  for (int t = wave; t < 27; t += 8) {         // 9 row-pairs x 3 col-tiles
    int row = 2 * ty + (m >> 3);
    int cp = 8 * tc + (m & 7); if (cp > 18) cp = 18;
    v8f c = {};
#pragma unroll
    for (int kc = 0; kc < 3; ++kc) {
      const _Float16* p0 = xs + ((row + ky0[kc]) * 21 + cp) * 8 + of0[kc];
      const _Float16* p1 = xs + ((row + ky1[kc]) * 21 + cp) * 8 + of1[kc];
      Q4 q; q.a = ldu64(p0); q.b = ldu64(p0 + 4); q.c = ldu64(p1); q.d = ldu64(p1 + 4);
      c = __builtin_amdgcn_wmma_f32_16x16x32_f16(false, q4v(q), false, bf[kc],
                                                 (short)0, c, false, false);
    }
#pragma unroll
    for (int r = 0; r < 8; ++r) scr[wave][r + 8 * half][col] = c[r];
#pragma unroll
    for (int it = 0; it < 2; ++it) {           // pxi in {half, half+2}
      int pxi = half + 2 * it;
      int px = 4 * tc + pxi;
      if (px < 9) {
        int m0 = 2 * pxi;
        float v0 = scr[wave][m0][col],     v1 = scr[wave][m0 + 1][col];
        float v2 = scr[wave][m0 + 8][col], v3 = scr[wave][m0 + 9][col];
        float mx = fmaxf(fmaxf(fmaxf(v0, v1), fmaxf(v2, v3)) + bias, 0.f);
        h2p[(ty * 9 + px) * 16 + col] = (_Float16)mx;  // h2[b][g][y][x][16]
      }
    }
    tc += 2; ty += 2; if (tc >= 3) { tc -= 3; ty += 1; }
  }
}

// ---------------- conv3 (Cin=16,k=3,Cout=32) + relu + maxpool2 ----------------
__global__ __launch_bounds__(256) void conv3_kernel(
    const _Float16* __restrict__ h2, const _Float16* __restrict__ w3t,
    const float* __restrict__ b3, _Float16* __restrict__ h3) {
  const int bg = blockIdx.x;
  const int b = bg / G, g = bg % G;
  const int tid = threadIdx.x, lane = tid & 31, wave = tid >> 5;
  const int col = lane & 15, half = lane >> 4, m = lane & 15;

  __shared__ _Float16 xs[9 * 9 * 16];          // [y][x][16] == h2 slice
  __shared__ _Float16 wB[32 * 160];
  __shared__ float scr[8][16][16];

  const unsigned long long* src = (const unsigned long long*)(h2 + ((size_t)b * G + g) * 1296);
  for (int i = tid; i < 324; i += 256) ((unsigned long long*)xs)[i] = src[i];
  for (int i = tid; i < 1280; i += 256)
    ((unsigned long long*)wB)[i] = ((const unsigned long long*)(w3t + g * 5120))[i];
  __syncthreads();

  if (wave < 6) {                              // 3 row-pairs x 2 co-tiles
    int ty = wave >> 1, ct = wave & 1;
    v16h bf[5];
    int ky0[5], of0[5], ky1[5], of1[5];
#pragma unroll
    for (int kc = 0; kc < 5; ++kc) {
      const _Float16* p = wB + (ct * 16 + col) * 160 + kc * 32 + 8 * half;
      Q4 q; q.a = ldu64(p); q.b = ldu64(p + 4); q.c = ldu64(p + 16); q.d = ldu64(p + 20);
      bf[kc] = q4v(q);
      int K0 = kc * 32 + 8 * half;
      ky0[kc] = K0 / 48; of0[kc] = K0 - ky0[kc] * 48;
      int K1 = K0 + 16;
      ky1[kc] = K1 / 48; of1[kc] = K1 - ky1[kc] * 48;
    }

    int row = 2 * ty + (m >> 3);
    int cp = (m & 7); if (cp > 6) cp = 6;
    v8f c = {};
#pragma unroll
    for (int kc = 0; kc < 5; ++kc) {
      const _Float16* p0 = xs + ((row + ky0[kc]) * 9 + cp) * 16 + of0[kc];
      const _Float16* p1 = xs + ((row + ky1[kc]) * 9 + cp) * 16 + of1[kc];
      Q4 q; q.a = ldu64(p0); q.b = ldu64(p0 + 4); q.c = ldu64(p1); q.d = ldu64(p1 + 4);
      c = __builtin_amdgcn_wmma_f32_16x16x32_f16(false, q4v(q), false, bf[kc],
                                                 (short)0, c, false, false);
    }
#pragma unroll
    for (int r = 0; r < 8; ++r) scr[wave][r + 8 * half][col] = c[r];

    const int co = ct * 16 + col;
    const float bias = b3[g * 32 + co];
    _Float16* h3p = h3 + ((size_t)g * 128 + b) * 288 + co * 9 + ty * 3;
#pragma unroll
    for (int it = 0; it < 2; ++it) {           // pxi in {half, half+2}, valid < 3
      int pxi = half + 2 * it;
      if (pxi < 3) {
        int m0 = 2 * pxi;
        float v0 = scr[wave][m0][col],     v1 = scr[wave][m0 + 1][col];
        float v2 = scr[wave][m0 + 8][col], v3 = scr[wave][m0 + 9][col];
        float mx = fmaxf(fmaxf(fmaxf(v0, v1), fmaxf(v2, v3)) + bias, 0.f);
        // h3[g][b][288], feature k = co*9 + py*3 + px (reference flatten order)
        h3p[pxi] = (_Float16)mx;
      }
    }
  }
}

// ---------------- heads: [128x288]@[288x64], fused @[64x2]+bias ----------------
__global__ __launch_bounds__(256) void head_kernel(
    const _Float16* __restrict__ h3, const _Float16* __restrict__ lw1t,
    const float* __restrict__ lw2, const float* __restrict__ lb2,
    float* __restrict__ out) {
  const int g = blockIdx.x;
  const int tid = threadIdx.x, lane = tid & 31, wave = tid >> 5;
  const int col = lane & 15, half = lane >> 4;

  __shared__ _Float16 Bs[64 * 288];            // [n][K]
  for (int i = tid; i < 2304; i += 256)
    ((unsigned long long*)Bs)[i] = ((const unsigned long long*)(lw1t + (size_t)g * 18432))[i];
  __syncthreads();

  const int mt = wave;                          // 8 waves * 16 batch rows
  const int brow = mt * 16 + (lane & 15);
  const _Float16* Ap = h3 + ((size_t)g * 128 + brow) * 288;

  v8f z = {};
  v8f c[4];
#pragma unroll
  for (int nt = 0; nt < 4; ++nt) c[nt] = z;

  for (int kc = 0; kc < 9; ++kc) {             // K = 288 = 9*32 exact
    const _Float16* pa = Ap + kc * 32 + 8 * half;
    Q4 qa; qa.a = ldu64(pa); qa.b = ldu64(pa + 4); qa.c = ldu64(pa + 16); qa.d = ldu64(pa + 20);
    v16h a = q4v(qa);
#pragma unroll
    for (int nt = 0; nt < 4; ++nt) {
      const _Float16* pb = Bs + (nt * 16 + col) * 288 + kc * 32 + 8 * half;
      Q4 qb; qb.a = ldu64(pb); qb.b = ldu64(pb + 4); qb.c = ldu64(pb + 16); qb.d = ldu64(pb + 20);
      c[nt] = __builtin_amdgcn_wmma_f32_16x16x32_f16(false, a, false, q4v(qb),
                                                     (short)0, c[nt], false, false);
    }
  }

  float a0[8], a1[8];
#pragma unroll
  for (int r = 0; r < 8; ++r) { a0[r] = 0.f; a1[r] = 0.f; }
#pragma unroll
  for (int nt = 0; nt < 4; ++nt) {
    int hh = nt * 16 + col;
    float w0 = lw2[(size_t)g * 128 + hh * 2];
    float w1v = lw2[(size_t)g * 128 + hh * 2 + 1];
#pragma unroll
    for (int r = 0; r < 8; ++r) { a0[r] += c[nt][r] * w0; a1[r] += c[nt][r] * w1v; }
  }
#pragma unroll
  for (int s = 1; s <= 8; s <<= 1) {
#pragma unroll
    for (int r = 0; r < 8; ++r) {
      a0[r] += __shfl_xor(a0[r], s);
      a1[r] += __shfl_xor(a1[r], s);
    }
  }
  if (col == 0) {
    float bb0 = lb2[g * 2], bb1 = lb2[g * 2 + 1];
#pragma unroll
    for (int r = 0; r < 8; ++r) {
      int b = mt * 16 + r + 8 * half;
      out[(size_t)b * 144 + g * 2]     = a0[r] + bb0;
      out[(size_t)b * 144 + g * 2 + 1] = a1[r] + bb1;
    }
  }
}

// ---------------- launch ----------------
extern "C" void kernel_launch(void* const* d_in, const int* in_sizes, int n_in,
                              void* d_out, int out_size, void* d_ws, size_t ws_size,
                              hipStream_t stream) {
  (void)in_sizes; (void)n_in; (void)out_size; (void)ws_size;
  const float* x   = (const float*)d_in[0];
  const float* w1  = (const float*)d_in[1];
  const float* b1  = (const float*)d_in[2];
  const float* w2  = (const float*)d_in[3];
  const float* b2  = (const float*)d_in[4];
  const float* w3  = (const float*)d_in[5];
  const float* b3  = (const float*)d_in[6];
  const float* lw1 = (const float*)d_in[7];
  const float* lw2 = (const float*)d_in[8];
  const float* lb2 = (const float*)d_in[9];

  char* ws = (char*)d_ws;                       // ~98.2 MB total
  _Float16* w1t  = (_Float16*)(ws + 0);           //   368,640 B
  _Float16* w2t  = (_Float16*)(ws + 368640);      //   221,184 B
  _Float16* w3t  = (_Float16*)(ws + 589824);      //   737,280 B
  _Float16* lw1t = (_Float16*)(ws + 1327104);     // 2,654,208 B
  _Float16* h1   = (_Float16*)(ws + 3981312);     // 65,028,096 B
  _Float16* h2   = (_Float16*)(ws + 69009408);    // 23,887,872 B
  _Float16* h3   = (_Float16*)(ws + 92897280);    //  5,308,416 B

  prep_w1 <<<(G * 16 * 160 + 255) / 256, 256, 0, stream>>>(w1, w1t);
  prep_w2 <<<(G * 16 * 96 + 255) / 256, 256, 0, stream>>>(w2, w2t);
  prep_w3 <<<(G * 32 * 160 + 255) / 256, 256, 0, stream>>>(w3, w3t);
  prep_lw1<<<(G * 64 * 288 + 255) / 256, 256, 0, stream>>>(lw1, lw1t);

  conv1_kernel<<<128 * G, 256, 0, stream>>>(x, w1t, b1, h1);
  conv2_kernel<<<128 * G, 256, 0, stream>>>(h1, w2t, b2, h2);
  conv3_kernel<<<128 * G, 256, 0, stream>>>(h2, w3t, b3, h3);
  head_kernel <<<G, 256, 0, stream>>>(h3, lw1t, lw2, lb2, (float*)d_out);
}
